// Classifier_32521492365559
// MI455X (gfx1250) — matrix operand order repaired and checked by reading, not compiled, not run on hardware
//
#include <hip/hip_runtime.h>
#include <hip/hip_bf16.h>

typedef __attribute__((ext_vector_type(16))) _Float16 v16h;
typedef __attribute__((ext_vector_type(8)))  float    v8f;

#define NNODES   50000
#define NEDGES   800000
#define FDIM     128
#define NGRAPHS  128
#define NCLASSES 10
#define MTILES   (NNODES / 16)   // 3125 exactly

// ---------------------------------------------------------------------------
// Degree / normalization
// ---------------------------------------------------------------------------
__global__ __launch_bounds__(256) void deg_init(float* __restrict__ deg) {
  int i = blockIdx.x * 256 + threadIdx.x;
  if (i < NNODES) deg[i] = 1.0f;               // self-loop contributes 1
}

__global__ __launch_bounds__(256) void deg_accum(const int* __restrict__ dst,
                                                 float* __restrict__ deg) {
  int e = blockIdx.x * 256 + threadIdx.x;
  if (e < NEDGES) unsafeAtomicAdd(&deg[dst[e]], 1.0f);
}

__global__ __launch_bounds__(256) void deg_invsqrt(float* __restrict__ deg) {
  int i = blockIdx.x * 256 + threadIdx.x;
  if (i < NNODES) deg[i] = rsqrtf(deg[i]);     // deg >= 1 always (self loop)
}

// ---------------------------------------------------------------------------
// Fused GCN-layer front half:
//   H    = act(X) @ W                       (v_wmma_f32_16x16x32_f16)
//   OUT  = dinv^2 * H + bias               (self-loop + bias init, fused
//                                           into the epilogue: saves a full
//                                           25.6MB read + write per layer)
// The edge scatter then accumulates the neighbor messages into OUT.
//  - W pre-swizzled into LDS in B-fragment order: 16 halves contiguous per
//    lane => 2x ds_load_b128 per fragment.
//  - A fragments: 4x global_load_b128 per k-step (ISA 16-bit A layout is
//    piecewise contiguous in K for a fixed lane-half).
// One wave per 16-row tile, covering all 8 column tiles.
// ---------------------------------------------------------------------------
template <bool RELU_IN>
__global__ __launch_bounds__(256) void gemm_wmma(const float* __restrict__ X,
                                                 const float* __restrict__ W,
                                                 const float* __restrict__ dinv,
                                                 const float* __restrict__ bias,
                                                 float* __restrict__ H,
                                                 float* __restrict__ OUT) {
  // Fragment-ordered W: [nt][ks][lane][e]  (8*4*32*16 halves = 32 KB)
  __shared__ _Float16 wfrag[8 * 4 * 32 * 16];
  const int tid = threadIdx.x;

  // Pre-swizzle: B layout is K = ks*32 + (e&7) + 16*(e>=8) + 8*(lane>=16), N = nt*16 + lane&15
  for (int j = tid; j < 8 * 4 * 32 * 16; j += 256) {
    int e  = j & 15;
    int ln = (j >> 4) & 31;
    int ks = (j >> 9) & 3;
    int nt = j >> 11;
    int k  = ks * 32 + (e & 7) + ((e >= 8) ? 16 : 0) + ((ln >= 16) ? 8 : 0);
    int n  = nt * 16 + (ln & 15);
    wfrag[j] = (_Float16)W[k * FDIM + n];
  }
  __syncthreads();

  const int wave  = tid >> 5;
  const int lane  = tid & 31;
  const int mtile = blockIdx.x * 8 + wave;
  if (mtile >= MTILES) return;                 // whole-wave uniform exit
  const int row0 = mtile * 16;
  const int m    = lane & 15;
  const int hi   = lane >> 4;                  // 0: lanes 0-15, 1: lanes 16-31

  // A fragments. ISA 16-bit A 16x32 layout: K = ks*32 + e + 8*(e>=8) + 8*hi.
  // For fixed hi: e=0..7 -> K=kbase..kbase+7, e=8..15 -> K=kbase+16..kbase+23,
  // where kbase = ks*32 + 8*hi  ==> 4 float4 loads per k-step.
  v16h a[4];
  const float* xrow = X + (size_t)(row0 + m) * FDIM;
  #pragma unroll
  for (int ks = 0; ks < 4; ++ks) {
    const int kbase = ks * 32 + hi * 8;
    float4 p0 = *(const float4*)(xrow + kbase);
    float4 p1 = *(const float4*)(xrow + kbase + 4);
    float4 p2 = *(const float4*)(xrow + kbase + 16);
    float4 p3 = *(const float4*)(xrow + kbase + 20);
    float t[16] = {p0.x, p0.y, p0.z, p0.w, p1.x, p1.y, p1.z, p1.w,
                   p2.x, p2.y, p2.z, p2.w, p3.x, p3.y, p3.z, p3.w};
    #pragma unroll
    for (int e = 0; e < 16; ++e) {
      float v = t[e];
      if (RELU_IN) v = v > 0.0f ? v : 0.0f;    // fold previous layer's ReLU
      a[ks][e] = (_Float16)v;
    }
  }

  // dinv^2 for this wave's 8 output rows (C/D layout: M = r + 8*hi)
  float di2[8];
  #pragma unroll
  for (int r = 0; r < 8; ++r) {
    float d = dinv[row0 + r + hi * 8];
    di2[r] = d * d;
  }

  const v16h* bfrag = (const v16h*)wfrag;      // 1024 fragments of 32 B
  for (int nt = 0; nt < 8; ++nt) {
    const int ncol = nt * 16 + (lane & 15);
    const float bcol = bias[ncol];
    v8f acc = {};
    #pragma unroll
    for (int ks = 0; ks < 4; ++ks) {
      v16h b = bfrag[(nt * 4 + ks) * 32 + lane];   // 2x ds_load_b128
      acc = __builtin_amdgcn_wmma_f32_16x16x32_f16(
          /*neg_a=*/false, a[ks], /*neg_b=*/false, b,
          /*c_mod=*/(short)0, acc, /*reuse_a=*/false, /*reuse_b=*/false);
    }
    #pragma unroll
    for (int r = 0; r < 8; ++r) {
      size_t idx = (size_t)(row0 + r + hi * 8) * FDIM + ncol;
      H[idx]   = acc[r];                       // h (gather source for scatter)
      OUT[idx] = di2[r] * acc[r] + bcol;       // self-loop + bias init
    }
  }
}

// ---------------------------------------------------------------------------
// Edge scatter: out[dst] += h[src] * dinv[src]*dinv[dst]. One wave per edge,
// 4 floats per lane (512-B coalesced gather), non-returning f32 atomics.
// Whole working set (~60 MB) is resident in the 192 MB L2.
// ---------------------------------------------------------------------------
__global__ __launch_bounds__(256) void edge_scatter(const int* __restrict__ src,
                                                    const int* __restrict__ dst,
                                                    const float* __restrict__ dinv,
                                                    const float* __restrict__ h,
                                                    float* __restrict__ out) {
  int e = blockIdx.x * 8 + (threadIdx.x >> 5);
  if (e >= NEDGES) return;
  int s = src[e];
  int d = dst[e];
  float nrm = dinv[s] * dinv[d];
  int lane = threadIdx.x & 31;
  const float4 v = *(const float4*)(h + (size_t)s * FDIM + lane * 4);
  float* o = out + (size_t)d * FDIM + lane * 4;
  unsafeAtomicAdd(o + 0, v.x * nrm);
  unsafeAtomicAdd(o + 1, v.y * nrm);
  unsafeAtomicAdd(o + 2, v.z * nrm);
  unsafeAtomicAdd(o + 3, v.w * nrm);
}

// ---------------------------------------------------------------------------
// Mean pool + final linear
// ---------------------------------------------------------------------------
__global__ __launch_bounds__(256) void pool_zero(float* __restrict__ sums,
                                                 float* __restrict__ cnts) {
  int i = blockIdx.x * 256 + threadIdx.x;
  if (i < NGRAPHS * FDIM) sums[i] = 0.0f;
  if (i < NGRAPHS) cnts[i] = 0.0f;
}

__global__ __launch_bounds__(256) void pool_accum(const float* __restrict__ h,
                                                  const int* __restrict__ bid,
                                                  float* __restrict__ sums,
                                                  float* __restrict__ cnts) {
  size_t idx = (size_t)blockIdx.x * 256 + threadIdx.x;   // N*128 threads
  int i = (int)(idx >> 7);
  int f = (int)(idx & 127);
  int g = bid[i];
  unsafeAtomicAdd(&sums[g * FDIM + f], h[idx]);
  if (f == 0) unsafeAtomicAdd(&cnts[g], 1.0f);
}

__global__ __launch_bounds__(256) void final_linear(const float* __restrict__ sums,
                                                    const float* __restrict__ cnts,
                                                    const float* __restrict__ lw,
                                                    const float* __restrict__ lb,
                                                    float* __restrict__ out) {
  int idx = blockIdx.x * 256 + threadIdx.x;
  if (idx >= NGRAPHS * NCLASSES) return;
  int g = idx / NCLASSES;
  int c = idx % NCLASSES;
  float inv = 1.0f / fmaxf(cnts[g], 1.0f);
  float acc = lb[c];
  #pragma unroll 8
  for (int k = 0; k < FDIM; ++k)
    acc += sums[g * FDIM + k] * inv * lw[k * NCLASSES + c];
  out[idx] = acc;
}

// ---------------------------------------------------------------------------
// Host launch
// ---------------------------------------------------------------------------
extern "C" void kernel_launch(void* const* d_in, const int* in_sizes, int n_in,
                              void* d_out, int out_size, void* d_ws, size_t ws_size,
                              hipStream_t stream) {
  const float* x     = (const float*)d_in[0];
  const int*   ei    = (const int*)d_in[1];     // (2, E): row0=src, row1=dst
  const int*   bid   = (const int*)d_in[2];
  const float* w1    = (const float*)d_in[3];
  const float* b1    = (const float*)d_in[4];
  const float* w2    = (const float*)d_in[5];
  const float* b2    = (const float*)d_in[6];
  const float* w3    = (const float*)d_in[7];
  const float* b3    = (const float*)d_in[8];
  const float* lin_w = (const float*)d_in[9];
  const float* lin_b = (const float*)d_in[10];
  float* out = (float*)d_out;

  const int* esrc = ei;
  const int* edst = ei + NEDGES;

  // Workspace layout (offsets 512B aligned)
  char* ws = (char*)d_ws;
  float* dinv = (float*)(ws);                                    // 50000 f
  float* bufA = (float*)(ws + 204800);                           // N*128 f
  float* bufB = (float*)(ws + 204800 + (size_t)NNODES*FDIM*4);   // N*128 f
  float* sums = (float*)(ws + 204800 + 2*(size_t)NNODES*FDIM*4); // 128*128 f
  float* cnts = (float*)(ws + 204800 + 2*(size_t)NNODES*FDIM*4 + NGRAPHS*FDIM*4);

  const int nodeBlocks = (NNODES + 255) / 256;
  const int edgeBlocks = (NEDGES + 255) / 256;
  const int featBlocks = (NNODES * FDIM) / 256;       // 25000
  const int gemmBlocks = (MTILES + 7) / 8;            // 391
  const int scatBlocks = (NEDGES + 7) / 8;            // 100000

  // 1) normalization
  deg_init<<<nodeBlocks, 256, 0, stream>>>(dinv);
  deg_accum<<<edgeBlocks, 256, 0, stream>>>(edst, dinv);
  deg_invsqrt<<<nodeBlocks, 256, 0, stream>>>(dinv);

  // 2) layer 1: bufA = x@w1, bufB = dinv^2*bufA + b1, then scatter into bufB
  gemm_wmma<false><<<gemmBlocks, 256, 0, stream>>>(x, w1, dinv, b1, bufA, bufB);
  edge_scatter<<<scatBlocks, 256, 0, stream>>>(esrc, edst, dinv, bufA, bufB);

  // 3) layer 2 (ReLU of bufB folded into A-load)
  gemm_wmma<true><<<gemmBlocks, 256, 0, stream>>>(bufB, w2, dinv, b2, bufA, bufB);
  edge_scatter<<<scatBlocks, 256, 0, stream>>>(esrc, edst, dinv, bufA, bufB);

  // 4) layer 3
  gemm_wmma<true><<<gemmBlocks, 256, 0, stream>>>(bufB, w3, dinv, b3, bufA, bufB);
  edge_scatter<<<scatBlocks, 256, 0, stream>>>(esrc, edst, dinv, bufA, bufB);

  // 5) mean pool over graphs + final linear
  pool_zero<<<(NGRAPHS * FDIM + 255) / 256, 256, 0, stream>>>(sums, cnts);
  pool_accum<<<featBlocks, 256, 0, stream>>>(bufB, bid, sums, cnts);
  final_linear<<<(NGRAPHS * NCLASSES + 255) / 256, 256, 0, stream>>>(
      sums, cnts, lin_w, lin_b, out);
}
